// SlotAttention_14757507629950
// MI455X (gfx1250) — matrix operand order repaired and checked by reading, not compiled, not run on hardware
//
#include <hip/hip_runtime.h>

// ---------------- problem constants ----------------
#define Bn  64
#define Nn  4096
#define Dn  128
#define Sn  7
#define SSn 64
#define Hn  128
#define LN_EPS 1e-5f
#define EPSA   1e-8f

typedef __bf16 bf16;
typedef __attribute__((ext_vector_type(4)))  bf16  v4bf;
typedef __attribute__((ext_vector_type(8)))  float v8f;
typedef __attribute__((ext_vector_type(8)))  bf16  v8bf;
typedef __attribute__((ext_vector_type(16))) bf16  v16bf;

// ---- CDNA5 WMMA: D(f32 16x16) = A(bf16 16x32) * B(bf16 32x16) + C ----
__device__ __forceinline__ v8f wmma_bf16(v16bf a, v16bf b, v8f c) {
  return __builtin_amdgcn_wmma_f32_16x16x32_bf16(false, a, false, b, (short)0, c,
                                                 false, false);
}

// Build a 16-half fragment from two 16-byte chunks (A-operand halves).
__device__ __forceinline__ v16bf frag16(const bf16* p0, const bf16* p1) {
  v8bf lo = *(const v8bf*)p0;
  v8bf hi = *(const v8bf*)p1;
  return __builtin_shufflevector(lo, hi, 0, 1, 2, 3, 4, 5, 6, 7,
                                 8, 9, 10, 11, 12, 13, 14, 15);
}

// ---- CDNA5 async global->LDS copy (ASYNCcnt path) ----
__device__ __forceinline__ void async_load_b128(unsigned lds_off, const void* g) {
  unsigned long long ga = (unsigned long long)(size_t)g;
  asm volatile("global_load_async_to_lds_b128 %0, %1, off"
               :: "v"(lds_off), "v"(ga) : "memory");
}
__device__ __forceinline__ void wait_async0() {
  asm volatile("s_wait_asynccnt 0x0" ::: "memory");
}
__device__ __forceinline__ void wait_ds0() {
  asm volatile("s_wait_dscnt 0x0" ::: "memory");
  __builtin_amdgcn_wave_barrier();
}

// =====================================================================
// k_init: slots = mu + exp(log_sigma)*slots_init ; wk/wv -> bf16
// =====================================================================
__global__ __launch_bounds__(256) void k_init(
    const float* __restrict__ slots_init, const float* __restrict__ mu,
    const float* __restrict__ lsig, const float* __restrict__ wk,
    const float* __restrict__ wv, float* __restrict__ slots,
    bf16* __restrict__ wkbf, bf16* __restrict__ wvbf) {
  int i = blockIdx.x * 256 + threadIdx.x;
  if (i < Bn * Sn * SSn) {
    int e = i & (SSn - 1);
    slots[i] = mu[e] + __expf(lsig[e]) * slots_init[i];
  }
  if (i < SSn * Dn) {
    wkbf[i] = (bf16)wk[i];
    wvbf[i] = (bf16)wv[i];
  }
}

// =====================================================================
// k_proj: fused LayerNorm(inputs) + k = x@wk^T, v = x@wv^T (bf16 WMMA).
// 4 waves/block, each wave owns a 16-row tile staged via async->LDS.
//  - k computed transposed-orientation (A=wk, B=xln) -> packed b128
//    stores into row-major k[n][e]
//  - v computed normal-orientation (A=xln, B=wv)     -> packed b128
//    stores into transposed vT[b][d][n]
// =====================================================================
__global__ __launch_bounds__(128) void k_proj(
    const float* __restrict__ x, const float* __restrict__ lnw,
    const float* __restrict__ lnb, const bf16* __restrict__ wkbf,
    const bf16* __restrict__ wvbf, bf16* __restrict__ kout,
    bf16* __restrict__ vTout) {
  __shared__ __align__(16) float s_lnw[Dn], s_lnb[Dn];
  __shared__ __align__(16) float s_x[4][16 * Dn];   // staged fp32 tiles
  __shared__ __align__(16) bf16  s_xb[4][16 * Dn];  // normalized bf16 tiles
  __shared__ float s_mean[4][16], s_rinv[4][16];

  int tid = threadIdx.x;
  if (tid < Dn) { s_lnw[tid] = lnw[tid]; s_lnb[tid] = lnb[tid]; }
  __syncthreads();

  int wave = tid >> 5, lane = tid & 31;
  int halfl = lane >> 4, l15 = lane & 15;
  long tile = (long)blockIdx.x * 4 + wave;  // 16-row tile over B*N rows
  long row0 = tile * 16;

  // 1) async copy 16x128 f32 tile into LDS (16 issues x 32 lanes x 16B)
  const float* gsrc = x + row0 * Dn;
  unsigned lbase = (unsigned)(size_t)(&s_x[wave][0]);
  for (int it = 0; it < 16; ++it) {
    async_load_b128(lbase + (unsigned)(it * 32 + lane) * 16u,
                    gsrc + (it * 32 + lane) * 4);
  }
  wait_async0();

  // 2) per-row LN stats: 2 lanes per row, vectorized ds_load_b128
  {
    int row = lane >> 1, half = lane & 1;
    const float4* r4 = (const float4*)&s_x[wave][row * Dn + half * 64];
    float sum = 0.f, sq = 0.f;
#pragma unroll
    for (int i = 0; i < 16; ++i) {
      float4 t = r4[i];
      sum += t.x + t.y + t.z + t.w;
      sq += t.x * t.x + t.y * t.y + t.z * t.z + t.w * t.w;
    }
    sum += __shfl_xor(sum, 1, 32);
    sq += __shfl_xor(sq, 1, 32);
    if (half == 0) {
      float m = sum * (1.f / Dn);
      s_mean[wave][row] = m;
      s_rinv[wave][row] = rsqrtf(sq * (1.f / Dn) - m * m + LN_EPS);
    }
  }
  wait_ds0();

  // 3) normalize + convert to bf16 (float4 in, v4bf/b64 out)
#pragma unroll
  for (int j = 0; j < 16; ++j) {
    int i4 = j * 32 + lane;            // float4 index, 512 per tile
    int rr = i4 >> 5;
    int cb = (i4 & 31) * 4;
    float4 t = ((const float4*)&s_x[wave][0])[i4];
    float m = s_mean[wave][rr], ri = s_rinv[wave][rr];
    v4bf pk;
    pk[0] = (bf16)((t.x - m) * ri * s_lnw[cb + 0] + s_lnb[cb + 0]);
    pk[1] = (bf16)((t.y - m) * ri * s_lnw[cb + 1] + s_lnb[cb + 1]);
    pk[2] = (bf16)((t.z - m) * ri * s_lnw[cb + 2] + s_lnb[cb + 2]);
    pk[3] = (bf16)((t.w - m) * ri * s_lnw[cb + 3] + s_lnb[cb + 3]);
    *(v4bf*)(&s_xb[wave][rr * Dn + cb]) = pk;
  }
  wait_ds0();

  // 4) WMMA. v-orientation A frags (xln rows) are loop invariant.
  const bf16* xb = &s_xb[wave][0];
  v16bf ax[4];
#pragma unroll
  for (int kk = 0; kk < 4; ++kk) {
    const bf16* p = xb + l15 * Dn + kk * 32 + halfl * 16;
    ax[kk] = frag16(p, p + 8);
  }

  long bb = row0 >> 12;           // batch index
  long nn = row0 & (Nn - 1);      // n within batch
  for (int c = 0; c < 4; ++c) {
    v8f acck = {0.f, 0.f, 0.f, 0.f, 0.f, 0.f, 0.f, 0.f};
    v8f accv = {0.f, 0.f, 0.f, 0.f, 0.f, 0.f, 0.f, 0.f};
#pragma unroll
    for (int kk = 0; kk < 4; ++kk) {
      // k-orientation: A = wk rows (e = c*16 + M), B = xln columns (n)
      const bf16* wkp = wkbf + (c * 16 + l15) * Dn + kk * 32 + halfl * 16;
      v16bf awk = frag16(wkp, wkp + 8);
      v16bf bx = *(const v16bf*)(xb + l15 * Dn + kk * 32 + halfl * 16);
      acck = wmma_bf16(awk, bx, acck);
      // v-orientation: A = xln rows (n), B = wv columns (e)
      v16bf bwv = *(const v16bf*)(wvbf + (c * 16 + l15) * Dn + kk * 32 +
                                  halfl * 16);
      accv = wmma_bf16(ax[kk], bwv, accv);
    }
    // k: lane holds (n = row0+l15, e = c*16 + halfl*8 + r) -> 1 b128 store
    v8bf pkk, pkv;
#pragma unroll
    for (int r = 0; r < 8; ++r) pkk[r] = (bf16)acck[r];
#pragma unroll
    for (int r = 0; r < 8; ++r) pkv[r] = (bf16)accv[r];
    *(v8bf*)(kout + (row0 + l15) * SSn + c * 16 + halfl * 8) = pkk;
    // v: lane holds (d = c*16+l15, n = row0 + halfl*8 + r) -> 1 b128 store
    *(v8bf*)(vTout + (bb * SSn + c * 16 + l15) * Nn + nn + halfl * 8) = pkv;
  }
}

// =====================================================================
// k_qproj (per batch): zero accumulators; q = LN(slots)@wq^T * SS^-0.5
// q stored bf16 [b][16][64], rows 7..15 zero-padded.
// =====================================================================
__global__ __launch_bounds__(64) void k_qproj(
    const float* __restrict__ slots, const float* __restrict__ wq,
    const float* __restrict__ lnw, const float* __restrict__ lnb,
    bf16* __restrict__ qbf, float* __restrict__ updates,
    float* __restrict__ colsum) {
  int b = blockIdx.x, t = threadIdx.x;
  __shared__ float sn[Sn][SSn];
  __shared__ float sm[Sn], sr[Sn];
  const float* sl = slots + b * Sn * SSn;

  for (int i = t; i < Sn * SSn; i += 64) updates[b * Sn * SSn + i] = 0.f;
  if (t < 16) colsum[b * 16 + t] = 0.f;

  if (t < Sn) {
    float sum = 0.f, sq = 0.f;
    for (int d = 0; d < SSn; ++d) { float v = sl[t * SSn + d]; sum += v; sq += v * v; }
    float m = sum * (1.f / SSn);
    sm[t] = m;
    sr[t] = rsqrtf(sq * (1.f / SSn) - m * m + LN_EPS);
  }
  __syncthreads();
  for (int s = 0; s < Sn; ++s)
    sn[s][t] = (sl[s * SSn + t] - sm[s]) * sr[s] * lnw[t] + lnb[t];
  __syncthreads();

  const float scale = 0.125f;  // SS^-0.5
  for (int s = 0; s < Sn; ++s) {
    float acc = 0.f;
    for (int d = 0; d < SSn; ++d) acc += sn[s][d] * wq[t * SSn + d];
    qbf[(b * 16 + s) * SSn + t] = (bf16)(acc * scale);
  }
  for (int s = Sn; s < 16; ++s) qbf[(b * 16 + s) * SSn + t] = (bf16)0.f;
}

// =====================================================================
// k_attn: logits^T = q@k^T (WMMA; slot logits land inside one lane ->
// in-register softmax) -> updates += attn^T @ v (WMMA over K=n).
// 2 blocks/batch, 4 waves/block, each wave owns 512 rows in 32-row steps.
// =====================================================================
__global__ __launch_bounds__(128) void k_attn(
    const bf16* __restrict__ kbf, const bf16* __restrict__ vT,
    const bf16* __restrict__ qbf, float* __restrict__ updates,
    float* __restrict__ colsum) {
  __shared__ __align__(16) bf16 s_att[4][16 * 32];

  int b = blockIdx.x >> 1, slice = blockIdx.x & 1;
  int tid = threadIdx.x, wave = tid >> 5, lane = tid & 31;
  int halfl = lane >> 4, l15 = lane & 15;

  // loop-invariant A frags of q (row s = l15, K = e), direct from global
  const bf16* qb = qbf + b * 16 * SSn;
  v16bf aq0 = frag16(qb + l15 * SSn + 0 + halfl * 16,
                     qb + l15 * SSn + 8 + halfl * 16);
  v16bf aq1 = frag16(qb + l15 * SSn + 32 + halfl * 16,
                     qb + l15 * SSn + 40 + halfl * 16);

  // attn staging tile: rows s (7..15 stay zero), cols = 32 local n
#pragma unroll
  for (int s = Sn; s < 16; ++s) s_att[wave][s * 32 + lane] = (bf16)0.f;

  float csum[Sn];
#pragma unroll
  for (int s = 0; s < Sn; ++s) csum[s] = 0.f;
  v8f upd[4];
#pragma unroll
  for (int c = 0; c < 4; ++c)
    upd[c] = (v8f){0.f, 0.f, 0.f, 0.f, 0.f, 0.f, 0.f, 0.f};

  const bf16* kb_b = kbf + (long)b * Nn * SSn;
  const bf16* vt_b = vT + (long)b * SSn * Nn;
  int nstart = slice * 2048 + wave * 512;

  for (int n0 = nstart; n0 < nstart + 512; n0 += 32) {
    __builtin_prefetch(kb_b + (long)(n0 + 32) * SSn, 0, 1);  // global_prefetch_b8
#pragma unroll
    for (int t = 0; t < 2; ++t) {
      // logits^T tile: C[s][n], n = n0 + t*16 + l15
      v8f lg = {0.f, 0.f, 0.f, 0.f, 0.f, 0.f, 0.f, 0.f};
      const bf16* kp = kb_b + (long)(n0 + t * 16 + l15) * SSn;
      v16bf bk0 = *(const v16bf*)(kp + 0 + halfl * 16);
      v16bf bk1 = *(const v16bf*)(kp + 32 + halfl * 16);
      lg = wmma_bf16(aq0, bk0, lg);
      lg = wmma_bf16(aq1, bk1, lg);

      // in-register softmax over s=0..6 (valid only for lanes 0..15)
      float vmax = -1e30f;
#pragma unroll
      for (int s = 0; s < Sn; ++s) vmax = fmaxf(vmax, lg[s]);
      float p[Sn], sum = 0.f;
#pragma unroll
      for (int s = 0; s < Sn; ++s) { p[s] = __expf(lg[s] - vmax); sum += p[s]; }
      float rinv = 1.f / sum;
      if (halfl == 0) {
#pragma unroll
        for (int s = 0; s < Sn; ++s) {
          float a = p[s] * rinv + EPSA;
          csum[s] += a;
          s_att[wave][s * 32 + t * 16 + l15] = (bf16)a;
        }
      }
    }
    wait_ds0();

    // updates += attn^T[16x32] @ v[32x16] per d-chunk (K = 32 rows)
    const bf16* ap = &s_att[wave][l15 * 32 + halfl * 16];
    v16bf aatt = frag16(ap, ap + 8);
#pragma unroll
    for (int c = 0; c < 4; ++c) {
      int d = c * 16 + l15;
      v16bf bv = *(const v16bf*)(vt_b + (long)d * Nn + n0 + halfl * 16);
      upd[c] = wmma_bf16(aatt, bv, upd[c]);
    }
  }

  // colsum: wave-reduce then one atomic per slot
#pragma unroll
  for (int s = 0; s < Sn; ++s) {
    float v = csum[s];
#pragma unroll
    for (int o = 16; o > 0; o >>= 1) v += __shfl_xor(v, o, 32);
    if (lane == 0) atomicAdd(&colsum[b * 16 + s], v);
  }
  // updates: scatter accumulators (s = r + 8*halfl, only s<7 valid)
#pragma unroll
  for (int c = 0; c < 4; ++c) {
#pragma unroll
    for (int r = 0; r < 8; ++r) {
      int s = r + halfl * 8;
      if (s < Sn)
        atomicAdd(&updates[(b * Sn + s) * SSn + c * 16 + l15], upd[c][r]);
    }
  }
}

// =====================================================================
// k_update (per batch): updates/=colsum; GRU; out = h + MLP(LN(h)).
// Writes slots workspace AND d_out (last iteration's value survives).
// =====================================================================
__global__ __launch_bounds__(64) void k_update(
    const float* __restrict__ updates, const float* __restrict__ colsum,
    float* __restrict__ slots, const float* __restrict__ wih,
    const float* __restrict__ whh, const float* __restrict__ bih,
    const float* __restrict__ bhh, const float* __restrict__ lnmw,
    const float* __restrict__ lnmb, const float* __restrict__ w1,
    const float* __restrict__ b1, const float* __restrict__ w2,
    const float* __restrict__ b2, float* __restrict__ dout) {
  int b = blockIdx.x, t = threadIdx.x;
  __shared__ float u[Sn][SSn], h[Sn][SSn], hn[Sn][SSn], hm[Sn][SSn];
  __shared__ float act[Sn][Hn];
  __shared__ float sm[Sn], sr[Sn];
  const float* up = updates + b * Sn * SSn;
  const float* cs = colsum + b * 16;
  float* sl = slots + b * Sn * SSn;

  for (int s = 0; s < Sn; ++s) {
    u[s][t] = up[s * SSn + t] / cs[s];
    h[s][t] = sl[s * SSn + t];
  }
  __syncthreads();

  for (int s = 0; s < Sn; ++s) {
    float gxr = bih[t], gxz = bih[SSn + t], gxn = bih[2 * SSn + t];
    float ghr = bhh[t], ghz = bhh[SSn + t], ghn = bhh[2 * SSn + t];
    for (int d = 0; d < SSn; ++d) {
      float ud = u[s][d], hd = h[s][d];
      gxr += ud * wih[t * SSn + d];
      gxz += ud * wih[(SSn + t) * SSn + d];
      gxn += ud * wih[(2 * SSn + t) * SSn + d];
      ghr += hd * whh[t * SSn + d];
      ghz += hd * whh[(SSn + t) * SSn + d];
      ghn += hd * whh[(2 * SSn + t) * SSn + d];
    }
    float r = 1.f / (1.f + __expf(-(gxr + ghr)));
    float z = 1.f / (1.f + __expf(-(gxz + ghz)));
    float n = tanhf(gxn + r * ghn);
    hn[s][t] = (1.f - z) * n + z * h[s][t];
  }
  __syncthreads();

  if (t < Sn) {
    float sum = 0.f, sq = 0.f;
    for (int d = 0; d < SSn; ++d) { float v = hn[t][d]; sum += v; sq += v * v; }
    float m = sum * (1.f / SSn);
    sm[t] = m;
    sr[t] = rsqrtf(sq * (1.f / SSn) - m * m + LN_EPS);
  }
  __syncthreads();
  for (int s = 0; s < Sn; ++s)
    hm[s][t] = (hn[s][t] - sm[s]) * sr[s] * lnmw[t] + lnmb[t];
  __syncthreads();

  for (int s = 0; s < Sn; ++s) {
    for (int jj = 0; jj < 2; ++jj) {
      int j = t + jj * 64;
      float a = b1[j];
      for (int d = 0; d < SSn; ++d) a += hm[s][d] * w1[j * SSn + d];
      act[s][j] = fmaxf(a, 0.f);
    }
  }
  __syncthreads();

  for (int s = 0; s < Sn; ++s) {
    float o = b2[t];
    for (int j = 0; j < Hn; ++j) o += act[s][j] * w2[t * Hn + j];
    float res = hn[s][t] + o;
    sl[s * SSn + t] = res;
    dout[b * Sn * SSn + s * SSn + t] = res;
  }
}

// =====================================================================
extern "C" void kernel_launch(void* const* d_in, const int* in_sizes, int n_in,
                              void* d_out, int out_size, void* d_ws,
                              size_t ws_size, hipStream_t stream) {
  (void)in_sizes; (void)n_in; (void)out_size; (void)ws_size;
  const float* inputs     = (const float*)d_in[0];
  const float* slots_init = (const float*)d_in[1];
  const float* ln_in_w    = (const float*)d_in[2];
  const float* ln_in_b    = (const float*)d_in[3];
  const float* ln_s_w     = (const float*)d_in[4];
  const float* ln_s_b     = (const float*)d_in[5];
  const float* ln_m_w     = (const float*)d_in[6];
  const float* ln_m_b     = (const float*)d_in[7];
  const float* wq         = (const float*)d_in[8];
  const float* wk         = (const float*)d_in[9];
  const float* wv         = (const float*)d_in[10];
  const float* gru_wih    = (const float*)d_in[11];
  const float* gru_whh    = (const float*)d_in[12];
  const float* gru_bih    = (const float*)d_in[13];
  const float* gru_bhh    = (const float*)d_in[14];
  const float* mlp_w1     = (const float*)d_in[15];
  const float* mlp_b1     = (const float*)d_in[16];
  const float* mlp_w2     = (const float*)d_in[17];
  const float* mlp_b2     = (const float*)d_in[18];
  const float* slot_mu    = (const float*)d_in[19];
  const float* slot_lsig  = (const float*)d_in[20];
  float* dout = (float*)d_out;

  char* ws = (char*)d_ws;
  size_t off = 0;
  auto carve = [&](size_t bytes) {
    void* p = ws + off;
    off = (off + bytes + 255) & ~(size_t)255;
    return p;
  };
  bf16*  kbf     = (bf16*)carve((size_t)Bn * Nn * SSn * 2);  // 33.5 MB
  bf16*  vT      = (bf16*)carve((size_t)Bn * SSn * Nn * 2);  // 33.5 MB
  bf16*  qbf     = (bf16*)carve((size_t)Bn * 16 * SSn * 2);
  bf16*  wkbf    = (bf16*)carve((size_t)SSn * Dn * 2);
  bf16*  wvbf    = (bf16*)carve((size_t)SSn * Dn * 2);
  float* slots   = (float*)carve((size_t)Bn * Sn * SSn * 4);
  float* updates = (float*)carve((size_t)Bn * Sn * SSn * 4);
  float* colsum  = (float*)carve((size_t)Bn * 16 * 4);

  k_init<<<(Bn * Sn * SSn + 255) / 256, 256, 0, stream>>>(
      slots_init, slot_mu, slot_lsig, wk, wv, slots, wkbf, wvbf);

  k_proj<<<(Bn * Nn) / (16 * 4), 128, 0, stream>>>(
      inputs, ln_in_w, ln_in_b, wkbf, wvbf, kbf, vT);

  for (int it = 0; it < 4; ++it) {  // ITERS=3 + stop-gradient step
    k_qproj<<<Bn, 64, 0, stream>>>(slots, wq, ln_s_w, ln_s_b, qbf, updates,
                                   colsum);
    k_attn<<<Bn * 2, 128, 0, stream>>>(kbf, vT, qbf, updates, colsum);
    k_update<<<Bn, 64, 0, stream>>>(updates, colsum, slots, gru_wih, gru_whh,
                                    gru_bih, gru_bhh, ln_m_w, ln_m_b, mlp_w1,
                                    mlp_b1, mlp_w2, mlp_b2, dout);
  }
}
